// SparseConv3DBase_17317308137881
// MI455X (gfx1250) — matrix operand order repaired and checked by reading, not compiled, not run on hardware
//
#include <hip/hip_runtime.h>

typedef __bf16 bf16_t;
typedef __attribute__((ext_vector_type(8)))  __bf16 v8bf;
typedef __attribute__((ext_vector_type(16))) __bf16 v16bf;
typedef __attribute__((ext_vector_type(8)))  float  v8f;

#define NROWS 200000
#define CIN   32
#define COUT  64
#define KV    27

// ---- fp32 -> bf16 (round-to-nearest-even) --------------------------------
__device__ __forceinline__ unsigned short f2bf(float f) {
    unsigned int u = __builtin_bit_cast(unsigned int, f);
    unsigned int r = u + 0x7FFFu + ((u >> 16) & 1u);
    return (unsigned short)(r >> 16);
}

// feats [NROWS][CIN] f32 -> bf16 [NROWS+1][CIN], row NROWS = zeros (pad row)
__global__ void convert_feats_kernel(const float* __restrict__ feats,
                                     unsigned short* __restrict__ out) {
    int i = blockIdx.x * blockDim.x + threadIdx.x;
    const int total = (NROWS + 1) * CIN;
    if (i >= total) return;
    float v = (i < NROWS * CIN) ? feats[i] : 0.0f;
    out[i] = f2bf(v);
}

// weight [KV][CIN][COUT] f32 -> bf16 transposed [KV][COUT][CIN]
__global__ void convert_weight_kernel(const float* __restrict__ w,
                                      unsigned short* __restrict__ wt) {
    int i = blockIdx.x * blockDim.x + threadIdx.x;
    const int total = KV * COUT * CIN;
    if (i >= total) return;
    int cin  = i % CIN;
    int cout = (i / CIN) % COUT;
    int k    = i / (CIN * COUT);
    wt[i] = f2bf(w[(k * CIN + cin) * COUT + cout]);
}

// ---- main kernel: one wave computes a 64-row x 64-cout output block ------
__global__ __launch_bounds__(256) void spconv_wmma_kernel(
    const bf16_t* __restrict__ feats_bf,   // [NROWS+1][CIN]
    const bf16_t* __restrict__ wt,         // [KV][COUT][CIN]
    const float*  __restrict__ bias,       // [COUT]
    const int*    __restrict__ in_idx,     // [KV][NROWS]
    float*        __restrict__ out)        // [NROWS][COUT]
{
    const int lane = threadIdx.x & 31;
    const int wave = threadIdx.x >> 5;
    const int gw   = blockIdx.x * 8 + wave;
    const int row0 = gw * 64;
    if (row0 >= NROWS) return;           // whole-wave uniform: EXEC stays all-1s

    const int m    = lane & 15;          // row/col within 16-wide tile
    const int half = lane >> 4;          // 0: lanes 0-15, 1: lanes 16-31

    // 16 accumulator tiles, bias folded into init (C layout: M=r+half*8, N=m)
    v8f c[4][4];
#pragma unroll
    for (int ct = 0; ct < 4; ++ct) {
        float bv = bias[ct * 16 + m];
#pragma unroll
        for (int rt = 0; rt < 4; ++rt)
#pragma unroll
            for (int r = 0; r < 8; ++r) c[rt][ct][r] = bv;
    }

#pragma unroll 1
    for (int k = 0; k < KV; ++k) {
        // B tiles: lane = column (cout), two cin chunks of 8 bf16.
        // B 32x16 layout: lanes 0-15 hold K=0-7 & 16-23, lanes 16-31 hold K=8-15 & 24-31.
        v16bf btile[4];
#pragma unroll
        for (int ct = 0; ct < 4; ++ct) {
            const bf16_t* bp = wt + ((k * COUT + ct * 16 + m) * CIN + half * 8);
            v8bf blo = *(const v8bf*)(bp);
            v8bf bhi = *(const v8bf*)(bp + 16);
            btile[ct] = __builtin_shufflevector(blo, bhi,
                0,1,2,3,4,5,6,7,8,9,10,11,12,13,14,15);
        }

        const int* idxp = in_idx + k * NROWS + row0 + m;
#pragma unroll
        for (int rt = 0; rt < 4; ++rt) {
            // Gather row via rulebook; pad index NROWS hits the zero row.
            int idx = idxp[rt * 16];
            // A 16x32 layout: lanes 0-15: K 0-7 (vgpr0-3) & 16-23 (vgpr4-7);
            //                 lanes 16-31: K 8-15 & 24-31.
            const bf16_t* ap = feats_bf + idx * CIN + half * 8;
            v8bf alo = *(const v8bf*)(ap);
            v8bf ahi = *(const v8bf*)(ap + 16);
            v16bf a = __builtin_shufflevector(alo, ahi,
                0,1,2,3,4,5,6,7,8,9,10,11,12,13,14,15);

#pragma unroll
            for (int ct = 0; ct < 4; ++ct) {
                c[rt][ct] = __builtin_amdgcn_wmma_f32_16x16x32_bf16(
                    false, a, false, btile[ct], (short)0, c[rt][ct],
                    false, false);
            }
        }
    }

    // Store: C/D tile layout -> row = rt*16 + half*8 + r, col = ct*16 + m
#pragma unroll
    for (int rt = 0; rt < 4; ++rt)
#pragma unroll
        for (int ct = 0; ct < 4; ++ct) {
            float* op = out + (row0 + rt * 16 + half * 8) * COUT + ct * 16 + m;
#pragma unroll
            for (int r = 0; r < 8; ++r) op[r * COUT] = c[rt][ct][r];
        }
}

extern "C" void kernel_launch(void* const* d_in, const int* in_sizes, int n_in,
                              void* d_out, int out_size, void* d_ws, size_t ws_size,
                              hipStream_t stream) {
    const float* feats  = (const float*)d_in[0];   // [NROWS][CIN]
    const float* weight = (const float*)d_in[1];   // [KV][CIN][COUT]
    const float* bias   = (const float*)d_in[2];   // [COUT]
    const int*   in_idx = (const int*)d_in[3];     // [KV][NROWS]
    float* out = (float*)d_out;

    // workspace layout: [wt_bf : KV*COUT*CIN bf16][feats_bf : (NROWS+1)*CIN bf16]
    unsigned short* wt_bf    = (unsigned short*)d_ws;
    unsigned short* feats_bf = wt_bf + (size_t)KV * COUT * CIN;

    {
        int total = KV * COUT * CIN;
        convert_weight_kernel<<<(total + 255) / 256, 256, 0, stream>>>(weight, wt_bf);
    }
    {
        int total = (NROWS + 1) * CIN;
        convert_feats_kernel<<<(total + 255) / 256, 256, 0, stream>>>(feats, feats_bf);
    }
    {
        int nwaves  = NROWS / 64;                 // 3125
        int nblocks = (nwaves + 7) / 8;           // 8 waves per block
        spconv_wmma_kernel<<<nblocks, 256, 0, stream>>>(
            (const bf16_t*)feats_bf, (const bf16_t*)wt_bf, bias, in_idx, out);
    }
}
